// SelfAttention_32469952758530
// MI455X (gfx1250) — matrix operand order repaired and checked by reading, not compiled, not run on hardware
//
#include <hip/hip_runtime.h>
#include <hip/hip_bf16.h>

typedef __attribute__((ext_vector_type(16))) _Float16 v16h;
typedef __attribute__((ext_vector_type(8)))  _Float16 v8h;
typedef __attribute__((ext_vector_type(8)))  float    v8f;
typedef __attribute__((ext_vector_type(4)))  unsigned v4u;
typedef __attribute__((ext_vector_type(8)))  int      v8i;
typedef __attribute__((ext_vector_type(4)))  int      v4i;

static constexpr int N_ = 4;
static constexpr int S_ = 2048;
static constexpr int E_ = 1024;
static constexpr int H_ = 16;
static constexpr int D_ = 64;               // head dim
static constexpr float SCALE_ = 0.03125f;   // 1/sqrt(E) = 1/32
static constexpr float NEGMASK_ = -1.0e20f * 0.03125f; // masked logit after scaling

__device__ __forceinline__ v16h cat8(v8h lo, v8h hi) {
  return __builtin_shufflevector(lo, hi, 0,1,2,3,4,5,6,7,8,9,10,11,12,13,14,15);
}

__device__ __forceinline__ v8f wmma_f16(v16h a, v16h b, v8f c) {
  // D = A(16x32 f16) * B(32x16 f16) + C(16x16 f32)
  return __builtin_amdgcn_wmma_f32_16x16x32_f16(false, a, false, b, (short)0, c,
                                                false, false);
}

// ---------------------------------------------------------------------------
// TDM: DMA a 2D tile (tile_w contiguous elems x tile_h rows, f16) from global
// memory into LDS. Descriptor per CDNA5 ISA D# layout.
// lds_off is the LDS byte offset (= low 32 bits of the generic pointer).
// ---------------------------------------------------------------------------
__device__ __forceinline__ void tdm_load_2d_f16(unsigned lds_off, const void* gptr,
                                                unsigned tile_w, unsigned tile_h,
                                                unsigned row_stride_elems) {
  unsigned long long ga = (unsigned long long)(uintptr_t)gptr;
  v4u g0;
  g0[0] = 1u;                                  // count=1, user descriptor
  g0[1] = lds_off;                             // lds_addr (bytes)
  g0[2] = (unsigned)(ga & 0xFFFFFFFFu);        // global_addr[31:0]
  g0[3] = (unsigned)((ga >> 32) & 0x01FFFFFFu) // global_addr[56:32]
          | (2u << 30);                        // type = 2 ("image")
  v8i g1;
  g1[0] = (int)(1u << 16);                     // data_size=1 -> 2 bytes/elem
  g1[1] = (int)((tile_w & 0xFFFFu) << 16);     // tensor_dim0[15:0] @ bits[63:48]
  g1[2] = (int)(((tile_w >> 16) & 0xFFFFu)     // tensor_dim0[31:16]
          | ((tile_h & 0xFFFFu) << 16));       // tensor_dim1[15:0]
  g1[3] = (int)(((tile_h >> 16) & 0xFFFFu)     // tensor_dim1[31:16]
          | ((tile_w & 0xFFFFu) << 16));       // tile_dim0
  g1[4] = (int)(tile_h & 0xFFFFu);             // tile_dim1 (tile_dim2 = 0)
  g1[5] = (int)row_stride_elems;               // tensor_dim0_stride[31:0]
  g1[6] = 0;
  g1[7] = 0;
  v4i gz = {0, 0, 0, 0};
#if __clang_major__ >= 23
  v8i gz8 = {0, 0, 0, 0, 0, 0, 0, 0};
  __builtin_amdgcn_tensor_load_to_lds(g0, g1, gz, gz, gz8, 0);
#else
  __builtin_amdgcn_tensor_load_to_lds(g0, g1, gz, gz, 0);
#endif
}

// ---------------------------------------------------------------------------
// Kernel 0: f32 -> f16 conversion (weights)
// ---------------------------------------------------------------------------
__global__ void cvt_f32_to_f16(const float* __restrict__ src,
                               _Float16* __restrict__ dst, int n) {
  int i = blockIdx.x * blockDim.x + threadIdx.x;
  if (i < n) dst[i] = (_Float16)src[i];
}

// ---------------------------------------------------------------------------
// Kernel 1: per-head projection  Y[s,e] = sum_d X[s, h*64+d] * W[e,d]
// One wave handles a 16-row (s) tile for one (n,h); 4 column tiles of 16.
// transposeOut: store Y transposed [e][s] (used for V so the P*V B-operand
// loads are contiguous).
// ---------------------------------------------------------------------------
__global__ __launch_bounds__(256) void proj_kernel(
    const float* __restrict__ X, const _Float16* __restrict__ W16,
    _Float16* __restrict__ Out, int transposeOut) {
  int lane = threadIdx.x & 31;
  int m = lane & 15, hi = lane >> 4;
  int wave = (int)((blockIdx.x * blockDim.x + threadIdx.x) >> 5);
  int st = wave & (S_ / 16 - 1);   // 128 s-tiles
  wave >>= 7;
  int h = wave & (H_ - 1);
  int n = wave >> 4;
  int s0 = st * 16;

  const float* xbase = X + (size_t)n * S_ * E_ + (size_t)h * D_;

  v8f c[4] = {};
  for (int d0 = 0; d0 < D_; d0 += 32) {
    // A operand: rows = s, K = d (f32 -> f16 on the fly)
    const float* p = xbase + (size_t)(s0 + m) * E_ + d0 + 16 * hi;
    v8f f0 = *(const v8f*)(p);
    v8f f1 = *(const v8f*)(p + 8);
    v16h a;
    #pragma unroll
    for (int i = 0; i < 8; ++i) { a[i] = (_Float16)f0[i]; a[8 + i] = (_Float16)f1[i]; }
    #pragma unroll
    for (int t = 0; t < 4; ++t) {
      // B[k=d][n=e] = W[e0+n][d0+k] : lane n holds W row (t*16+n), contiguous
      v16h b = *(const v16h*)(W16 + (size_t)(t * 16 + m) * D_ + d0 + 16 * hi);
      c[t] = wmma_f16(a, b, c[t]);
    }
  }

  size_t head = (size_t)(n * H_ + h);
  if (!transposeOut) {
    _Float16* outp = Out + head * S_ * D_;   // [s][d]
    #pragma unroll
    for (int t = 0; t < 4; ++t)
      #pragma unroll
      for (int r = 0; r < 8; ++r) {
        int row = s0 + r + 8 * hi, col = t * 16 + m;
        outp[(size_t)row * D_ + col] = (_Float16)c[t][r];
      }
  } else {
    _Float16* outp = Out + head * D_ * S_;   // [d][s]
    #pragma unroll
    for (int t = 0; t < 4; ++t)
      #pragma unroll
      for (int r = 0; r < 8; ++r) {
        int row = s0 + r + 8 * hi, col = t * 16 + m;
        outp[(size_t)col * S_ + row] = (_Float16)c[t][r];
      }
  }
}

// ---------------------------------------------------------------------------
// Kernel 2: flash attention. One block (8 waves) = one (n,h) and 128 query
// rows; each wave owns a 16-row q-tile. K (32x64) and V^T (64x32) tiles are
// DMA'd into LDS by the Tensor Data Mover, double buffered, shared by all 8
// waves (8x less L2 traffic than per-wave streaming).
// Qp, Kp: [n][h][s][d] f16.  Vpt: [n][h][d][s] f16.  mask: [n][1][S][S] i32.
// Writes Xattn: [n*S][E] f16 (rows = query, cols = h*64+d).
// ---------------------------------------------------------------------------
__global__ __launch_bounds__(256) void attn_kernel(
    const _Float16* __restrict__ Qp, const _Float16* __restrict__ Kp,
    const _Float16* __restrict__ Vpt, const int* __restrict__ mask,
    _Float16* __restrict__ Xattn) {
  __shared__ __align__(32) _Float16 Kt[2][32 * 64];   // [buf][key][d]
  __shared__ __align__(32) _Float16 Vt[2][64 * 32];   // [buf][d][key]
  __shared__ __align__(32) _Float16 Pst[8][16 * 40];  // per-wave P tile, padded

  int wid = threadIdx.x >> 5;
  int lane = threadIdx.x & 31;
  int m = lane & 15, hi = lane >> 4;
  int blk = blockIdx.x;            // N*H*(S/128) = 1024
  int qb = (blk & 15) * 128;
  int h = (blk >> 4) & 15;
  int n = blk >> 8;
  int q0 = qb + wid * 16;

  const _Float16* Qb = Qp + (size_t)(n * H_ + h) * S_ * D_;
  const _Float16* Kb = Kp + (size_t)(n * H_ + h) * S_ * D_;
  const _Float16* Vb = Vpt + (size_t)(n * H_ + h) * D_ * S_;
  const int* mb = mask + (size_t)n * S_ * S_;

  // Q tile (16 x 64) as two A operands, kept in registers for the whole loop.
  const _Float16* qp = Qb + (size_t)(q0 + m) * D_ + 16 * hi;
  v16h aq0 = cat8(*(const v8h*)(qp), *(const v8h*)(qp + 8));        // d 0..31
  v16h aq1 = cat8(*(const v8h*)(qp + 32), *(const v8h*)(qp + 40));  // d 32..63

  float mrow[8], lrow[8];
  v8f o[4] = {};
  #pragma unroll
  for (int r = 0; r < 8; ++r) { mrow[r] = -3.0e38f; lrow[r] = 0.0f; }

  // Prologue: TDM tile 0 into buffer 0 (wave 0 issues; TDM ignores EXEC).
  if (wid == 0) {
    tdm_load_2d_f16((unsigned)(uintptr_t)&Kt[0][0], Kb, 64, 32, 64);
    tdm_load_2d_f16((unsigned)(uintptr_t)&Vt[0][0], Vb, 32, 64, S_);
  }

  for (int it = 0; it < S_ / 32; ++it) {
    int kb = it * 32;
    int buf = it & 1;
    if (wid == 0) {
      if (it + 1 < S_ / 32) {
        // issue next tile into the other buffer, then wait for current tile
        tdm_load_2d_f16((unsigned)(uintptr_t)&Kt[buf ^ 1][0],
                        Kb + (size_t)(kb + 32) * D_, 64, 32, 64);
        tdm_load_2d_f16((unsigned)(uintptr_t)&Vt[buf ^ 1][0],
                        Vb + (kb + 32), 32, 64, S_);
        __builtin_amdgcn_s_wait_tensorcnt(2);
      } else {
        __builtin_amdgcn_s_wait_tensorcnt(0);
      }
    }
    __syncthreads();  // publish current buffer to all waves

    const _Float16* KtB = &Kt[buf][0];
    const _Float16* VtB = &Vt[buf][0];

    // --- QK^T for two 16-key tiles, K-chained over D=64 (B from LDS) ---
    v16h bk0a = *(const v16h*)(KtB + m * 64 + 16 * hi);
    v16h bk0b = *(const v16h*)(KtB + m * 64 + 32 + 16 * hi);
    v16h bk1a = *(const v16h*)(KtB + (16 + m) * 64 + 16 * hi);
    v16h bk1b = *(const v16h*)(KtB + (16 + m) * 64 + 32 + 16 * hi);
    v8f s0t = {}, s1t = {};
    s0t = wmma_f16(aq0, bk0a, s0t);
    s0t = wmma_f16(aq1, bk0b, s0t);
    s1t = wmma_f16(aq0, bk1a, s1t);
    s1t = wmma_f16(aq1, bk1b, s1t);

    // --- mask, scale, online softmax (rows live in C-layout slots) ---
    #pragma unroll
    for (int r = 0; r < 8; ++r) {
      int qrow = q0 + r + 8 * hi;
      const int* mrp = mb + (size_t)qrow * S_ + kb + m;
      float e0 = mrp[0]  ? s0t[r] * SCALE_ : NEGMASK_;
      float e1 = mrp[16] ? s1t[r] * SCALE_ : NEGMASK_;
      float mx = fmaxf(e0, e1);
      #pragma unroll
      for (int off = 8; off >= 1; off >>= 1) mx = fmaxf(mx, __shfl_xor(mx, off, 32));
      float mnew = fmaxf(mrow[r], mx);
      float alpha = __expf(mrow[r] - mnew);
      mrow[r] = mnew;
      float p0 = __expf(e0 - mnew);
      float p1 = __expf(e1 - mnew);
      float rs = p0 + p1;
      #pragma unroll
      for (int off = 8; off >= 1; off >>= 1) rs += __shfl_xor(rs, off, 32);
      lrow[r] = lrow[r] * alpha + rs;
      #pragma unroll
      for (int t = 0; t < 4; ++t) o[t][r] *= alpha;
      // stage P (f16) to LDS in row-major [q][key], padded stride 40
      Pst[wid][(r + 8 * hi) * 40 + m] = (_Float16)p0;
      Pst[wid][(r + 8 * hi) * 40 + 16 + m] = (_Float16)p1;
    }
    asm volatile("s_wait_dscnt 0x0" ::: "memory");  // DS wave-ordered; wait stores
    __builtin_amdgcn_wave_barrier();

    // --- P (A operand from LDS) x V^T (B operand from LDS) ---
    const _Float16* lp = &Pst[wid][0] + m * 40 + 16 * hi;
    v16h ap = cat8(*(const v8h*)(lp), *(const v8h*)(lp + 8));
    #pragma unroll
    for (int t = 0; t < 4; ++t) {
      v16h bv = *(const v16h*)(VtB + (t * 16 + m) * 32 + 16 * hi);
      o[t] = wmma_f16(ap, bv, o[t]);
    }
    __syncthreads();  // all waves done with buf before TDM overwrites it
  }

  // --- normalize and store as f16 rows of Xattn ---
  #pragma unroll
  for (int r = 0; r < 8; ++r) {
    float inv = 1.0f / lrow[r];
    int row = q0 + r + 8 * hi;
    #pragma unroll
    for (int t = 0; t < 4; ++t) {
      int col = h * D_ + t * 16 + m;
      Xattn[((size_t)(n * S_ + row)) * E_ + col] = (_Float16)(o[t][r] * inv);
    }
  }
}

// ---------------------------------------------------------------------------
// Kernel 3: final projection  Out[i][j] = sum_e X[i][e] * Wo[j][e] + bo[j]
// 16x64 output tile per wave, K-loop of 32 steps (16x16x32 WMMA).
// ---------------------------------------------------------------------------
__global__ __launch_bounds__(256) void outproj_kernel(
    const _Float16* __restrict__ X, const _Float16* __restrict__ Wo16,
    const float* __restrict__ bo, float* __restrict__ Out) {
  int lane = threadIdx.x & 31;
  int m = lane & 15, hi = lane >> 4;
  int wave = blockIdx.x * 8 + (threadIdx.x >> 5);
  int jt = wave & 15;       // 16 column tiles of 64
  int it = wave >> 4;       // 512 row tiles of 16
  int i0 = it * 16, j0 = jt * 64;

  v8f c[4] = {};
  for (int k0 = 0; k0 < E_; k0 += 32) {
    const _Float16* p = X + (size_t)(i0 + m) * E_ + k0 + 16 * hi;
    v16h a = cat8(*(const v8h*)(p), *(const v8h*)(p + 8));
    #pragma unroll
    for (int t = 0; t < 4; ++t) {
      v16h b = *(const v16h*)(Wo16 + (size_t)(j0 + t * 16 + m) * E_ + k0 + 16 * hi);
      c[t] = wmma_f16(a, b, c[t]);
    }
  }
  #pragma unroll
  for (int t = 0; t < 4; ++t) {
    float bias = bo[j0 + t * 16 + m];
    #pragma unroll
    for (int r = 0; r < 8; ++r) {
      int row = i0 + r + 8 * hi;
      Out[(size_t)row * E_ + j0 + t * 16 + m] = c[t][r] + bias;
    }
  }
}

// ---------------------------------------------------------------------------
extern "C" void kernel_launch(void* const* d_in, const int* in_sizes, int n_in,
                              void* d_out, int out_size, void* d_ws, size_t ws_size,
                              hipStream_t stream) {
  (void)in_sizes; (void)n_in; (void)out_size; (void)ws_size;
  const float* values = (const float*)d_in[0];
  const float* keys   = (const float*)d_in[1];
  const float* query  = (const float*)d_in[2];
  const int*   mask   = (const int*)d_in[3];
  const float* Wv     = (const float*)d_in[4];
  const float* Wk     = (const float*)d_in[5];
  const float* Wq     = (const float*)d_in[6];
  const float* Wo     = (const float*)d_in[7];
  const float* bo     = (const float*)d_in[8];
  float* out = (float*)d_out;

  _Float16* ws = (_Float16*)d_ws;
  const size_t szNHSD = (size_t)N_ * H_ * S_ * D_;  // 8M halves
  _Float16* Vpt  = ws;                  // [n][h][d][s]
  _Float16* Kp   = ws + szNHSD;         // [n][h][s][d]
  _Float16* Qp   = ws + 2 * szNHSD;     // [n][h][s][d]
  _Float16* Xa   = ws + 3 * szNHSD;     // [n*S][E]
  _Float16* Wo16 = ws + 4 * szNHSD;
  _Float16* Wv16 = Wo16 + (size_t)E_ * E_;
  _Float16* Wk16 = Wv16 + D_ * D_;
  _Float16* Wq16 = Wk16 + D_ * D_;

  cvt_f32_to_f16<<<(E_ * E_ + 255) / 256, 256, 0, stream>>>(Wo, Wo16, E_ * E_);
  cvt_f32_to_f16<<<(D_ * D_ + 255) / 256, 256, 0, stream>>>(Wv, Wv16, D_ * D_);
  cvt_f32_to_f16<<<(D_ * D_ + 255) / 256, 256, 0, stream>>>(Wk, Wk16, D_ * D_);
  cvt_f32_to_f16<<<(D_ * D_ + 255) / 256, 256, 0, stream>>>(Wq, Wq16, D_ * D_);

  // N*H*(S/16) = 8192 waves, 8 waves per 256-thread block
  proj_kernel<<<1024, 256, 0, stream>>>(values, Wv16, Vpt, 1);
  proj_kernel<<<1024, 256, 0, stream>>>(keys,   Wk16, Kp,  0);
  proj_kernel<<<1024, 256, 0, stream>>>(query,  Wq16, Qp,  0);

  // N*H*(S/128) = 1024 blocks, TDM-fed flash attention
  attn_kernel<<<1024, 256, 0, stream>>>(Qp, Kp, Vpt, mask, Xa);

  // (N*S/16) * (E/64) = 8192 waves
  outproj_kernel<<<1024, 256, 0, stream>>>(Xa, Wo16, bo, out);
}
